// KMeansClusteringLoss_44281112821850
// MI455X (gfx1250) — matrix operand order repaired and checked by reading, not compiled, not run on hardware
//
#include <hip/hip_runtime.h>
#include <hip/hip_bf16.h>
#include <math.h>

// ---------------------------------------------------------------------------
// KMeans min-distance loss on MI455X (gfx1250), wave32 + WMMA bf16 split-hi/lo
//   d2[l,c] = |z_l|^2 + |mu_c|^2 - 2 z_l . mu_c ;  out = mean_l min_c sqrt(d2)
// Cross term via V_WMMA_F32_16X16X32_BF16 with A = bf16hi/lo(-2*z), B = bf16
// hi/lo(mu) staged in LDS (entire mu fits: 2*139KB + 2KB norms < 320KB/WGP).
// Round-2 change: hoist all B-operand LDS loads (and m2) ahead of the 12
// WMMAs per centroid tile -> one s_wait_dscnt per iteration instead of four.
// ---------------------------------------------------------------------------

typedef __attribute__((ext_vector_type(16))) __bf16 bf16x16;
typedef __attribute__((ext_vector_type(8)))  __bf16 bf16x8;
typedef __attribute__((ext_vector_type(8)))  float  f32x8;

#define DIMK   128      // feature dim
#define NCENT  512      // centroids
#define NPTS   65536    // points
#define LDS_STRIDE 136  // bf16 elems per LDS row (128 + 8 pad -> 4-dword bank skew)
#define WAVES_PER_BLOCK 8
#define NBLOCKS 512     // 512 blocks * 8 waves = 4096 row-tiles of 16 points

constexpr int    MUH_ELEMS = NCENT * LDS_STRIDE;
constexpr size_t LDS_BYTES = (size_t)2 * MUH_ELEMS * sizeof(__bf16)  // mu hi + lo
                           + NCENT * sizeof(float)                   // |mu|^2
                           + WAVES_PER_BLOCK * sizeof(float);        // wave sums

__global__ __launch_bounds__(256, 1)
void kmeans_min_dist_kernel(const float* __restrict__ z,
                            const float* __restrict__ mu,
                            float* __restrict__ partials)
{
    extern __shared__ char smem[];
    __bf16* muh  = (__bf16*)smem;                                   // hi part
    __bf16* mul  = muh + MUH_ELEMS;                                 // lo part
    float*  m2   = (float*)(smem + (size_t)2 * MUH_ELEMS * sizeof(__bf16));
    float*  wsum = m2 + NCENT;

    const int t = threadIdx.x;

    // ---- Stage mu into LDS as split bf16 + exact fp32 row norms (2 rows/thr).
    for (int r = t; r < NCENT; r += blockDim.x) {
        const float* mp = mu + r * DIMK;
        float s2 = 0.f;
        #pragma unroll 4
        for (int k = 0; k < DIMK; k += 4) {
            float4 f = *(const float4*)(mp + k);
            float xs[4] = {f.x, f.y, f.z, f.w};
            #pragma unroll
            for (int i = 0; i < 4; ++i) {
                float x = xs[i];
                s2 += x * x;                       // fixed order -> deterministic
                __bf16 h = (__bf16)x;
                muh[r * LDS_STRIDE + k + i] = h;
                mul[r * LDS_STRIDE + k + i] = (__bf16)(x - (float)h);
            }
        }
        m2[r] = s2;
    }
    __syncthreads();

    const int w    = t >> 5;
    const int lane = t & 31;
    const int l16  = lane & 15;   // A row / B column / D column within tile
    const int kh   = lane >> 4;   // K-half selector per ISA 16-bit layouts

    float wacc = 0.f;
    const int numTiles = NPTS / 16;

    for (int tile = blockIdx.x * WAVES_PER_BLOCK + w; tile < numTiles;
         tile += gridDim.x * WAVES_PER_BLOCK)
    {
        const int row = tile * 16 + l16;
        const float* zp = z + (size_t)row * DIMK;

        // ---- Load z row, fold -2 scale, split to bf16 hi/lo in A layout:
        // lane(kh) holds K in {kb*32+kh*8 .. +7} U {kb*32+16+kh*8 .. +7}.
        bf16x16 ah[4], al[4];
        float z2p = 0.f;
        #pragma unroll
        for (int kb = 0; kb < 4; ++kb) {
            const int k0 = kb * 32 + kh * 8;
            float4 f0 = *(const float4*)(zp + k0);
            float4 f1 = *(const float4*)(zp + k0 + 4);
            float4 f2 = *(const float4*)(zp + k0 + 16);
            float4 f3 = *(const float4*)(zp + k0 + 20);
            float xs[16] = {f0.x, f0.y, f0.z, f0.w, f1.x, f1.y, f1.z, f1.w,
                            f2.x, f2.y, f2.z, f2.w, f3.x, f3.y, f3.z, f3.w};
            #pragma unroll
            for (int e = 0; e < 16; ++e) {
                float x = xs[e];
                z2p += x * x;                      // exact |z|^2 partial (fp32)
                float y = -2.0f * x;
                __bf16 h = (__bf16)y;
                ah[kb][e] = h;
                al[kb][e] = (__bf16)(y - (float)h);
            }
        }

        float smin[8];
        #pragma unroll
        for (int r = 0; r < 8; ++r) smin[r] = __builtin_inff();

        // ---- Sweep all 512 centroids (32 N-tiles) out of LDS.
        for (int nt = 0; nt < 32; ++nt) {
            const int c = nt * 16 + l16;
            const bf16x8* bhp = (const bf16x8*)(muh + c * LDS_STRIDE);
            const bf16x8* blp = (const bf16x8*)(mul + c * LDS_STRIDE);

            // Issue ALL LDS traffic for this tile first (16x ds_load_b128 +
            // one b32), so a single dscnt wait covers the whole WMMA burst.
            bf16x16 bh[4], bl[4];
            #pragma unroll
            for (int kb = 0; kb < 4; ++kb) {
                bh[kb] = __builtin_shufflevector(bhp[kb * 4 + kh], bhp[kb * 4 + 2 + kh],
                    0, 1, 2, 3, 4, 5, 6, 7, 8, 9, 10, 11, 12, 13, 14, 15);
                bl[kb] = __builtin_shufflevector(blp[kb * 4 + kh], blp[kb * 4 + 2 + kh],
                    0, 1, 2, 3, 4, 5, 6, 7, 8, 9, 10, 11, 12, 13, 14, 15);
            }
            const float m2c = m2[c];

            f32x8 acc0 = {};   // hi*hi chain
            f32x8 acc1 = {};   // cross-term chain (independent -> ILP of 2)
            #pragma unroll
            for (int kb = 0; kb < 4; ++kb) {
                acc0 = __builtin_amdgcn_wmma_f32_16x16x32_bf16(
                    false, ah[kb], false, bh[kb], (short)0, acc0, false, false);
                acc1 = __builtin_amdgcn_wmma_f32_16x16x32_bf16(
                    false, ah[kb], false, bl[kb], (short)0, acc1, false, false);
                acc1 = __builtin_amdgcn_wmma_f32_16x16x32_bf16(
                    false, al[kb], false, bh[kb], (short)0, acc1, false, false);
            }

            #pragma unroll
            for (int r = 0; r < 8; ++r) {
                float s = acc0[r] + acc1[r] + m2c;   // |mu|^2 - 2 z.mu
                smin[r] = fminf(smin[r], s);
            }
        }

        // ---- Min across the 16 D-columns held by each lane half.
        #pragma unroll
        for (int m = 1; m < 16; m <<= 1) {
            #pragma unroll
            for (int r = 0; r < 8; ++r)
                smin[r] = fminf(smin[r], __shfl_xor(smin[r], m, 32));
        }
        // lane 0 now holds rows m0+0..7 ; lane 16 rows m0+8..15.
        float rowmin = 0.f;
        #pragma unroll
        for (int r = 0; r < 8; ++r) {
            float v0 = __shfl(smin[r], 0, 32);
            float v1 = __shfl(smin[r], 16, 32);
            if (lane == r)     rowmin = v0;
            if (lane == r + 8) rowmin = v1;
        }

        // Full |z|^2 for row m0+l16 (lane L + lane L+16 cover complementary K).
        float z2 = z2p + __shfl_xor(z2p, 16, 32);
        float dist = (lane < 16) ? sqrtf(fmaxf(z2 + rowmin, 0.f)) : 0.f;

        // Deterministic wave sum of the 16 distances.
        #pragma unroll
        for (int m = 1; m < 32; m <<= 1) dist += __shfl_xor(dist, m, 32);
        if (lane == 0) wacc += dist;
    }

    if (lane == 0) wsum[w] = wacc;
    __syncthreads();
    if (t == 0) {
        float s = 0.f;
        #pragma unroll
        for (int i = 0; i < WAVES_PER_BLOCK; ++i) s += wsum[i];
        partials[blockIdx.x] = s;
    }
}

// Fixed-order final reduction -> deterministic scalar mean.
__global__ void kmeans_finalize_kernel(const float* __restrict__ partials,
                                       float* __restrict__ out,
                                       int n, float inv)
{
    if (threadIdx.x == 0 && blockIdx.x == 0) {
        float s = 0.f;
        for (int i = 0; i < n; ++i) s += partials[i];
        out[0] = s * inv;
    }
}

extern "C" void kernel_launch(void* const* d_in, const int* in_sizes, int n_in,
                              void* d_out, int out_size, void* d_ws, size_t ws_size,
                              hipStream_t stream)
{
    (void)in_sizes; (void)n_in; (void)out_size; (void)ws_size;
    const float* z  = (const float*)d_in[0];   // (65536, 1, 128) fp32
    const float* mu = (const float*)d_in[1];   // (512, 128) fp32
    float* out      = (float*)d_out;           // scalar fp32
    float* partials = (float*)d_ws;            // NBLOCKS floats of scratch

    // 281 KB dynamic LDS (> default cap) — raise the attribute every call
    // (idempotent host-side call; no static guards, graph-capture safe).
    (void)hipFuncSetAttribute((const void*)kmeans_min_dist_kernel,
                              hipFuncAttributeMaxDynamicSharedMemorySize,
                              (int)LDS_BYTES);

    kmeans_min_dist_kernel<<<NBLOCKS, 256, LDS_BYTES, stream>>>(z, mu, partials);
    kmeans_finalize_kernel<<<1, 32, 0, stream>>>(partials, out, NBLOCKS,
                                                 1.0f / (float)NPTS);
}
